// GraphSAGE_58506044506625
// MI455X (gfx1250) — matrix operand order repaired and checked by reading, not compiled, not run on hardware
//
#include <hip/hip_runtime.h>
#include <hip/hip_bf16.h>

typedef __attribute__((ext_vector_type(16))) __bf16 v16bf;
typedef __attribute__((ext_vector_type(8)))  __bf16 v8bf;
typedef __attribute__((ext_vector_type(8)))  float  v8f;

// async-to-LDS builtin operand types: v2i ptr in AS(1) (global) / AS(3) (LDS)
typedef int v2ia __attribute__((vector_size(8)));
typedef __attribute__((address_space(1))) v2ia* as1_v2i;
typedef __attribute__((address_space(3))) v2ia* as3_v2i;

#define NN 100000
#define NE 1600000
#define FIN 128
#define KDIM 256   // concatenated [h ; h_neigh]

#if __has_builtin(__builtin_amdgcn_global_load_async_to_lds_b64)
#define HAVE_ASYNC_LDS 1
#else
#define HAVE_ASYNC_LDS 0
#endif

// ---------------- utility ----------------
__global__ void zero_f32(float* __restrict__ p, int n) {
  int i = blockIdx.x * blockDim.x + threadIdx.x;
  int stride = gridDim.x * blockDim.x;
  for (; i < n; i += stride) p[i] = 0.0f;
}

// in-degree (depends only on dst; computed once per launch)
__global__ void degree_kernel(const int* __restrict__ dst, float* __restrict__ deg, int nE) {
  int e = blockIdx.x * blockDim.x + threadIdx.x;
  if (e < nE) atomicAdd(&deg[dst[e]], 1.0f);
}

// Build Wcat[n][k] (bf16, n=output col padded to Fpad, k=0..255 over [W_self;W_neigh])
// i.e. transpose + concat + f32->bf16, plus padded bias.
__global__ void prep_w(const float* __restrict__ Ws, const float* __restrict__ Wn,
                       const float* __restrict__ b, __bf16* __restrict__ Wcat,
                       float* __restrict__ bpad, int Fout, int Fpad) {
  int i = blockIdx.x * blockDim.x + threadIdx.x;
  int total = Fpad * KDIM;
  if (i < total) {
    int n = i / KDIM, k = i % KDIM;
    float v = 0.0f;
    if (n < Fout) v = (k < FIN) ? Ws[k * Fout + n] : Wn[(k - FIN) * Fout + n];
    Wcat[i] = (__bf16)v;
  }
  if (i < Fpad) bpad[i] = (i < Fout) ? b[i] : 0.0f;
}

// Edge aggregation: one wave per edge (uniform e -> scalar index loads),
// 32 lanes x float4, f32 atomics (everything L2-resident: h is 51MB < 192MB L2).
__global__ void agg_kernel(const float* __restrict__ h, const int* __restrict__ src,
                           const int* __restrict__ dst, float* __restrict__ sum) {
  long t = (long)blockIdx.x * blockDim.x + threadIdx.x;
  long e = t >> 5;
  int lane = (int)(t & 31);
  if (e >= NE) return;
  // e is wave-uniform; make it provably so -> s_load of src/dst, saddr-form VMEM
  int eu = __builtin_amdgcn_readfirstlane((int)e);
  int s = src[eu], d = dst[eu];
  const float4 v = *(const float4*)(h + (long)s * FIN + lane * 4);
  float* o = sum + (long)d * FIN + lane * 4;
  atomicAdd(o + 0, v.x); atomicAdd(o + 1, v.y);
  atomicAdd(o + 2, v.z); atomicAdd(o + 3, v.w);
}

// Fused SAGE layer GEMM: OUT[16-row tile] = [h ; sum/deg] (16x256) @ Wcat^T + bias (+ReLU)
// One wave per 16x16 output tile; 8x v_wmma_f32_16x16x32_bf16 over K=256.
// W_self half (K<128) of this wave's weight slice is DMA'd to LDS with
// GLOBAL_LOAD_ASYNC_TO_LDS (overlapped with A-tile f32->bf16 staging), synced
// with s_wait_asynccnt; W_neigh half stays on direct global_load_b128.
__global__ void sage_gemm(const float* __restrict__ H, const float* __restrict__ SUM,
                          const float* __restrict__ DEG, const __bf16* __restrict__ Wcat,
                          const float* __restrict__ bpad, float* __restrict__ OUT,
                          int FoutStore, int doRelu) {
  __shared__ __align__(32) __bf16 Alds[16 * KDIM];       // 8 KB
  __shared__ __align__(32) __bf16 Wlds[8 * 16 * FIN];    // 32 KB (K-lo slice, 8 waves)
  const int rbase = blockIdx.x * 16;

  const int wave = threadIdx.x >> 5;
  const int lane = threadIdx.x & 31;
  const int n0 = wave * 16;

#if HAVE_ASYNC_LDS
  // Kick off async DMA of this wave's 16x128 (K<128) bf16 weight slice into LDS.
  {
    __bf16* wdst0 = &Wlds[(wave * 16) * FIN];
#pragma unroll
    for (int row = 0; row < 16; ++row) {
      const __bf16* gsrc = Wcat + (long)(n0 + row) * KDIM + lane * 4;  // 8B per lane
      __bf16* ldst = wdst0 + row * FIN + lane * 4;
      __builtin_amdgcn_global_load_async_to_lds_b64(
          (as1_v2i)(v2ia*)(uintptr_t)gsrc, (as3_v2i)(v2ia*)ldst, 0, 0);
    }
  }
#endif

  // Stage A tile into LDS as bf16: cols 0..127 = h, 128..255 = sum * 1/max(deg,1)
  // (VALU conversion work overlaps with the async weight DMA above)
  for (int s = threadIdx.x; s < 16 * 16; s += blockDim.x) {
    int row = s >> 4, c = s & 15;
    long node = rbase + row;
    float scale = 1.0f;
    const float* srcp;
    if (c < 8) {
      srcp = H + node * FIN + c * 16;
    } else {
      srcp = SUM + node * FIN + (c - 8) * 16;
      scale = 1.0f / fmaxf(DEG[node], 1.0f);
    }
    __bf16* outp = Alds + row * KDIM + c * 16;
#pragma unroll
    for (int j = 0; j < 16; ++j) outp[j] = (__bf16)(srcp[j] * scale);
  }
  __syncthreads();
#if HAVE_ASYNC_LDS
#if __has_builtin(__builtin_amdgcn_s_wait_asynccnt)
  __builtin_amdgcn_s_wait_asynccnt(0);
#else
  asm volatile("s_wait_asynccnt 0x0" ::: "memory");
#endif
#endif

  const int half = lane >> 4;        // 0: lanes 0-15, 1: lanes 16-31
  const int arow = lane & 15;        // A: M = lane%16
  const int khalfA = half * 16;      // A lanes0-15: K 0..15; lanes16-31: K 16..31
  const int koffB = half * 16;       // B lanes0-15: K 0..15; lanes16-31: K 16..31
  const int ncol = lane & 15;        // B/D: N = lane%16

  v8f acc = {};
  const __bf16* wrow    = Wcat + (long)(n0 + ncol) * KDIM;
  const __bf16* wldsrow = &Wlds[(wave * 16 + ncol) * FIN];
  const __bf16* arowp   = Alds + arow * KDIM;
#pragma unroll
  for (int kc = 0; kc < KDIM / 32; ++kc) {
    v8bf alo = *(const v8bf*)(arowp + kc * 32 + khalfA);
    v8bf ahi = *(const v8bf*)(arowp + kc * 32 + 8 + khalfA);
    v16bf a;
#pragma unroll
    for (int j = 0; j < 8; ++j) { a[j] = alo[j]; a[j + 8] = ahi[j]; }
    v16bf bm;
#if HAVE_ASYNC_LDS
    if (kc < 4) bm = *(const v16bf*)(wldsrow + kc * 32 + koffB);   // LDS (async-staged)
    else        bm = *(const v16bf*)(wrow + kc * 32 + koffB);      // global (L2-hot)
#else
    bm = *(const v16bf*)(wrow + kc * 32 + koffB);
#endif
    acc = __builtin_amdgcn_wmma_f32_16x16x32_bf16(false, a, false, bm, (short)0, acc,
                                                  false, false);
  }

  const int col = n0 + ncol;
  const float bia = bpad[col];
  if (col < FoutStore) {
#pragma unroll
    for (int i = 0; i < 8; ++i) {
      int m = i + half * 8;                 // D layout: VGPR i -> M = i + 8*(lane/16)
      long node = rbase + m;
      float v = acc[i] + bia;
      if (doRelu) v = fmaxf(v, 0.0f);
      OUT[node * FoutStore + col] = v;
    }
  }
}

// ---------------- launcher ----------------
extern "C" void kernel_launch(void* const* d_in, const int* in_sizes, int n_in,
                              void* d_out, int out_size, void* d_ws, size_t ws_size,
                              hipStream_t stream) {
  (void)in_sizes; (void)n_in; (void)out_size; (void)ws_size;
  const float* X   = (const float*)d_in[0];
  const int*   src = (const int*)d_in[1];
  const int*   dst = (const int*)d_in[2];
  const float* Ws0 = (const float*)d_in[3];
  const float* Wn0 = (const float*)d_in[4];
  const float* b0  = (const float*)d_in[5];
  const float* Ws1 = (const float*)d_in[6];
  const float* Wn1 = (const float*)d_in[7];
  const float* b1  = (const float*)d_in[8];
  const float* Ws2 = (const float*)d_in[9];
  const float* Wn2 = (const float*)d_in[10];
  const float* b2  = (const float*)d_in[11];

  char* ws = (char*)d_ws;
  auto alloc = [&](size_t bytes) {
    char* p = ws; ws += (bytes + 255) & ~(size_t)255; return p;
  };
  float*  sum = (float*)alloc((size_t)NN * FIN * 4);
  float*  h1  = (float*)alloc((size_t)NN * FIN * 4);
  float*  h2  = (float*)alloc((size_t)NN * FIN * 4);
  float*  deg = (float*)alloc((size_t)NN * 4);
  __bf16* W0  = (__bf16*)alloc((size_t)128 * KDIM * 2);
  __bf16* W1  = (__bf16*)alloc((size_t)128 * KDIM * 2);
  __bf16* W2  = (__bf16*)alloc((size_t)48 * KDIM * 2);
  float*  b0p = (float*)alloc(128 * 4);
  float*  b1p = (float*)alloc(128 * 4);
  float*  b2p = (float*)alloc(48 * 4);

  // degree + weight prep (once per launch; deterministic)
  zero_f32<<<512, 256, 0, stream>>>(deg, NN);
  degree_kernel<<<(NE + 255) / 256, 256, 0, stream>>>(dst, deg, NE);
  prep_w<<<(128 * KDIM + 255) / 256, 256, 0, stream>>>(Ws0, Wn0, b0, W0, b0p, 128, 128);
  prep_w<<<(128 * KDIM + 255) / 256, 256, 0, stream>>>(Ws1, Wn1, b1, W1, b1p, 128, 128);
  prep_w<<<(48 * KDIM + 255) / 256, 256, 0, stream>>>(Ws2, Wn2, b2, W2, b2p, 40, 48);

  const int aggBlocks = (int)(((long)NE * 32) / 256);  // 200000
  const int rowTiles  = NN / 16;                       // 6250

  // layer 0
  zero_f32<<<2048, 256, 0, stream>>>(sum, NN * FIN);
  agg_kernel<<<aggBlocks, 256, 0, stream>>>(X, src, dst, sum);
  sage_gemm<<<rowTiles, 256, 0, stream>>>(X, sum, deg, W0, b0p, h1, 128, 1);
  // layer 1
  zero_f32<<<2048, 256, 0, stream>>>(sum, NN * FIN);
  agg_kernel<<<aggBlocks, 256, 0, stream>>>(h1, src, dst, sum);
  sage_gemm<<<rowTiles, 256, 0, stream>>>(h1, sum, deg, W1, b1p, h2, 128, 1);
  // layer 2 (Fpad=48 -> 3 waves of 32; store guard col<40, no ReLU)
  zero_f32<<<2048, 256, 0, stream>>>(sum, NN * FIN);
  agg_kernel<<<aggBlocks, 256, 0, stream>>>(h2, src, dst, sum);
  sage_gemm<<<rowTiles, 96, 0, stream>>>(h2, sum, deg, W2, b2p, (float*)d_out, 40, 0);
}
